// RNNAttentionDecoder_85100482003419
// MI455X (gfx1250) — matrix operand order repaired and checked by reading, not compiled, not run on hardware
//
#include <hip/hip_runtime.h>
#include <hip/hip_bf16.h>
#include <math.h>

// ---------------------------------------------------------------------------
// Problem constants (B,S,H,A,M,V,E) = (128,64,1000,1000,500,30000,620)
// ---------------------------------------------------------------------------
constexpr int cB = 128, cS = 64, cH = 1000, cA = 1000, cM = 500, cV = 30000, cE = 620;

typedef __bf16 bf16_t;
typedef __attribute__((ext_vector_type(16))) __bf16 v16bf;
typedef __attribute__((ext_vector_type(8)))  __bf16 v8bf;
typedef __attribute__((ext_vector_type(8)))  float  v8f;

// ---------------------------------------------------------------------------
// WMMA bf16 GEMM:  C[M,N] = A[M,Kp](bf16, zero-padded) * B[N,Kp]^T + bias
// One wave per 32(M) x 64(N) macro-tile: 2x4 = 8 accumulators of 16x16.
// Per K-step (K+=32): 4 A-loads + 4 B-loads feed 8 v_wmma_f32_16x16x32_bf16
// (4x A-frag reuse, 2x B-frag reuse).
// Fragment packing (this code's contiguous k order, k = 16*(lane/16)+s, 16-bit operands, wave32):
//   A: lane holds row M=lane%16; lanes 0-15 K=[0..15],
//      lanes 16-31 K=[16..31]  -> two contiguous 16B loads.
//   B: lane holds column N=lane%16; lanes 0-15 K=[0..15], 16-31 K=[16..31]
//      -> one contiguous 32B load (weights are (N,K) row-major = B^T columns).
//   C/D: VGPR r -> M = r + 8*(lane/16), N = lane%16.
// Requirements honored: Kp % 32 == 0, M % 32 == 0, gridDim.x*64 <= Np (padded).
// ---------------------------------------------------------------------------
__global__ __launch_bounds__(32)
void gemm_bf16_wmma(const bf16_t* __restrict__ Am, const bf16_t* __restrict__ Bm,
                    const float* __restrict__ bias, float* __restrict__ C,
                    int M, int N, int Kp, int ldc)
{
    const int lane = threadIdx.x & 31;
    const int half = lane >> 4;       // 0 or 1
    const int l16  = lane & 15;
    const int m0 = blockIdx.y << 5;   // 32 rows per wave
    const int n0 = blockIdx.x << 6;   // 64 cols per wave

    const bf16_t* Ar0 = Am + (long)(m0 + l16) * Kp + half * 16;
    const bf16_t* Ar1 = Ar0 + (long)16 * Kp;
    const bf16_t* Br[4];
    #pragma unroll
    for (int j = 0; j < 4; ++j)
        Br[j] = Bm + (long)(n0 + j * 16 + l16) * Kp + half * 16;

    v8f acc[2][4];
    #pragma unroll
    for (int i = 0; i < 2; ++i)
        #pragma unroll
        for (int j = 0; j < 4; ++j)
            acc[i][j] = (v8f){};

    for (int k = 0; k < Kp; k += 32) {
        __builtin_prefetch(Ar0 + k + 512, 0, 0);   // -> global_prefetch_b8
        __builtin_prefetch(Ar1 + k + 512, 0, 0);
        __builtin_prefetch(Br[0] + k + 512, 0, 0);
        __builtin_prefetch(Br[2] + k + 512, 0, 0);

        union { v16bf v; v8bf h[2]; } a0, a1;
        a0.h[0] = *(const v8bf*)(Ar0 + k);
        a0.h[1] = *(const v8bf*)(Ar0 + k + 8);
        a1.h[0] = *(const v8bf*)(Ar1 + k);
        a1.h[1] = *(const v8bf*)(Ar1 + k + 8);

        #pragma unroll
        for (int j = 0; j < 4; ++j) {
            v16bf bf = *(const v16bf*)(Br[j] + k);
            acc[0][j] = __builtin_amdgcn_wmma_f32_16x16x32_bf16(
                            false, a0.v, false, bf, (short)0, acc[0][j], false, false);
            acc[1][j] = __builtin_amdgcn_wmma_f32_16x16x32_bf16(
                            false, a1.v, false, bf, (short)0, acc[1][j], false, false);
        }
    }

    #pragma unroll
    for (int j = 0; j < 4; ++j) {
        const int n = n0 + j * 16 + l16;
        if (n < N) {
            const float bv = bias ? bias[n] : 0.0f;
            #pragma unroll
            for (int i = 0; i < 2; ++i) {
                #pragma unroll
                for (int r = 0; r < 8; ++r) {
                    const int m = m0 + i * 16 + r + half * 8;   // M % 32 == 0 here
                    C[(long)m * ldc + n] = acc[i][j][r] + bv;
                }
            }
        }
    }
}

// ---------------------------------------------------------------------------
// f32 -> zero-padded bf16 (full padded extent, writes zeros in pad region)
// ---------------------------------------------------------------------------
__global__ void cvt_pad(const float* __restrict__ src, bf16_t* __restrict__ dst,
                        int N, int K, int Kp, long total)
{
    long i = (long)blockIdx.x * blockDim.x + threadIdx.x;
    if (i >= total) return;
    int n = (int)(i / Kp), k = (int)(i % Kp);
    float v = (k < K && n < N) ? src[(long)n * K + k] : 0.0f;
    dst[i] = (bf16_t)v;
}

// f32 segment -> bf16 sub-block at column offset (for concatenated operands)
__global__ void cvt_seg(const float* __restrict__ src, bf16_t* __restrict__ dst,
                        int rows, int cols, int src_ld, int dst_ld, int col0)
{
    long i = (long)blockIdx.x * blockDim.x + threadIdx.x;
    if (i >= (long)rows * cols) return;
    int r = (int)(i / cols), c = (int)(i % cols);
    dst[(long)r * dst_ld + col0 + c] = (bf16_t)src[(long)r * src_ld + c];
}

__global__ void zero_bf16(bf16_t* __restrict__ p, long n)
{
    long i = (long)blockIdx.x * blockDim.x + threadIdx.x;
    if (i < n) p[i] = (bf16_t)0.0f;
}

// emb_x[b,:] = emb[x[b],:]
__global__ void gather_emb(const int* __restrict__ x, const float* __restrict__ emb,
                           float* __restrict__ emb_x)
{
    int b = blockIdx.x;
    int idx = x[b];
    for (int e = threadIdx.x; e < cE; e += blockDim.x)
        emb_x[(long)b * cE + e] = emb[(long)idx * cE + e];
}

__global__ void bias3_k(const float* a, const float* b, const float* c,
                        float* o, int n)
{
    int i = blockIdx.x * blockDim.x + threadIdx.x;
    if (i < n) o[i] = a[i] + b[i] + c[i];
}

// energy[b,s] = tanh(h_proj[b,:] + o_proj[b,s,:]) . v_e
__global__ __launch_bounds__(256)
void energy_k(const float* __restrict__ h_proj, const float* __restrict__ o_proj,
              const float* __restrict__ v_e, float* __restrict__ energy)
{
    __shared__ float red[256];
    int bs = blockIdx.x;
    int b  = bs / cS;
    const float* hp = h_proj + (long)b  * cA;
    const float* op = o_proj + (long)bs * cA;
    float s = 0.0f;
    for (int a = threadIdx.x; a < cA; a += 256)
        s += tanhf(hp[a] + op[a]) * v_e[a];
    red[threadIdx.x] = s;
    __syncthreads();
    for (int off = 128; off > 0; off >>= 1) {
        if (threadIdx.x < off) red[threadIdx.x] += red[threadIdx.x + off];
        __syncthreads();
    }
    if (threadIdx.x == 0) energy[bs] = red[0];
}

// softmax over S (per batch row) + ctx[b,k] = sum_s attn[b,s]*outputs[b,s,k]
__global__ __launch_bounds__(256)
void attn_ctx_k(const float* __restrict__ energy, const float* __restrict__ outputs,
                float* __restrict__ ctx)
{
    __shared__ float attn[cS];
    int b = blockIdx.x;
    if (threadIdx.x < cS) attn[threadIdx.x] = energy[b * cS + threadIdx.x];
    __syncthreads();
    if (threadIdx.x == 0) {
        float mx = attn[0];
        for (int s = 1; s < cS; ++s) mx = fmaxf(mx, attn[s]);
        float sum = 0.0f;
        for (int s = 0; s < cS; ++s) { float e = expf(attn[s] - mx); attn[s] = e; sum += e; }
        float inv = 1.0f / sum;
        for (int s = 0; s < cS; ++s) attn[s] *= inv;
    }
    __syncthreads();
    const float* ob = outputs + (long)b * cS * (2 * cH);
    for (int k = threadIdx.x; k < 2 * cH; k += 256) {
        float acc = 0.0f;
        for (int s = 0; s < cS; ++s)
            acc += attn[s] * ob[(long)s * (2 * cH) + k];
        ctx[(long)b * (2 * cH) + k] = acc;
    }
}

// GRU gates -> h_new
__global__ void gru_k(const float* __restrict__ gi, const float* __restrict__ gh,
                      const float* __restrict__ hidden, float* __restrict__ h_new)
{
    long i = (long)blockIdx.x * blockDim.x + threadIdx.x;
    if (i >= (long)cB * cH) return;
    int b = (int)(i / cH), j = (int)(i % cH);
    const float* gib = gi + (long)b * 3 * cH;
    const float* ghb = gh + (long)b * 3 * cH;
    float r = 1.0f / (1.0f + expf(-(gib[j]          + ghb[j])));
    float z = 1.0f / (1.0f + expf(-(gib[cH + j]     + ghb[cH + j])));
    float n = tanhf(gib[2 * cH + j] + r * ghb[2 * cH + j]);
    h_new[i] = (1.0f - z) * n + z * hidden[i];
}

// maxout pairs -> padded bf16 (128 x 512), pad columns zeroed
__global__ void maxout_k(const float* __restrict__ mo_in, bf16_t* __restrict__ mo_bf)
{
    long i = (long)blockIdx.x * blockDim.x + threadIdx.x;
    if (i >= (long)cB * 512) return;
    int b = (int)(i / 512), m = (int)(i % 512);
    float v = 0.0f;
    if (m < cM)
        v = fmaxf(mo_in[(long)b * 2 * cM + 2 * m], mo_in[(long)b * 2 * cM + 2 * m + 1]);
    mo_bf[i] = (bf16_t)v;
}

// ---------------------------------------------------------------------------
extern "C" void kernel_launch(void* const* d_in, const int* in_sizes, int n_in,
                              void* d_out, int out_size, void* d_ws, size_t ws_size,
                              hipStream_t stream)
{
    const int*   x       = (const int*)  d_in[0];
    const float* hidden  = (const float*)d_in[1];
    const float* outputs = (const float*)d_in[2];
    const float* emb     = (const float*)d_in[3];
    const float* W_hp    = (const float*)d_in[4];
    const float* b_hp    = (const float*)d_in[5];
    const float* W_op    = (const float*)d_in[6];
    const float* b_op    = (const float*)d_in[7];
    const float* v_e     = (const float*)d_in[8];
    const float* W_ih    = (const float*)d_in[9];
    const float* W_hh    = (const float*)d_in[10];
    const float* b_ih    = (const float*)d_in[11];
    const float* b_hh    = (const float*)d_in[12];
    const float* W_wv    = (const float*)d_in[13];
    const float* b_wv    = (const float*)d_in[14];
    const float* W_lh    = (const float*)d_in[15];
    const float* b_lh    = (const float*)d_in[16];
    const float* W_le    = (const float*)d_in[17];
    const float* b_le    = (const float*)d_in[18];
    const float* W_pred  = (const float*)d_in[19];
    const float* b_pred  = (const float*)d_in[20];

    float* pred  = (float*)d_out;                       // (128, 30000)
    float* h_new = (float*)d_out + (long)cB * cV;       // (128, 1000)

    // ---- workspace carving (aligned) ----
    size_t off = 0;
    auto carve = [&](size_t bytes) -> char* {
        char* p = (char*)d_ws + off;
        off += (bytes + 255) & ~size_t(255);
        return p;
    };
    // padded bf16 operands (N padded to x64 for the 64-wide macro-tile)
    bf16_t* Whp_bf   = (bf16_t*)carve((size_t)1024 * 1024 * 2);     // (1000,1000)->(1024,1024)
    bf16_t* Wop_bf   = (bf16_t*)carve((size_t)1024 * 2016 * 2);     // (1000,2000)->(1024,2016)
    bf16_t* Wih_bf   = (bf16_t*)carve((size_t)3008 * 2624 * 2);     // (3000,2620)->(3008,2624)
    bf16_t* Whh_bf   = (bf16_t*)carve((size_t)3008 * 1024 * 2);     // (3000,1000)->(3008,1024)
    bf16_t* Wcat_bf  = (bf16_t*)carve((size_t)1024 * 3648 * 2);     // [W_lh|W_le|W_wv] K=3620->3648
    bf16_t* Wpred_bf = (bf16_t*)carve((size_t)30016 * 512 * 2);     // (30000,500)->(30016,512)
    bf16_t* hid_bf   = (bf16_t*)carve((size_t)cB * 1024 * 2);       // (128,1000)->(128,1024)
    bf16_t* out_bf   = (bf16_t*)carve((size_t)cB * cS * 2016 * 2);  // (8192,2000)->(8192,2016)
    bf16_t* gcat_bf  = (bf16_t*)carve((size_t)cB * 2624 * 2);       // [emb_x|ctx] K=2620->2624
    bf16_t* acat_bf  = (bf16_t*)carve((size_t)cB * 3648 * 2);       // [hidden|emb_x|ctx]
    bf16_t* mo_bf    = (bf16_t*)carve((size_t)cB * 512 * 2);        // (128,500)->(128,512)
    // f32 intermediates
    float* h_proj = (float*)carve((size_t)cB * cA * 4);
    float* o_proj = (float*)carve((size_t)cB * cS * cA * 4);
    float* energy = (float*)carve((size_t)cB * cS * 4);
    float* ctx    = (float*)carve((size_t)cB * 2 * cH * 4);
    float* emb_x  = (float*)carve((size_t)cB * cE * 4);
    float* gi     = (float*)carve((size_t)cB * 3 * cH * 4);
    float* gh     = (float*)carve((size_t)cB * 3 * cH * 4);
    float* mo_in  = (float*)carve((size_t)cB * 2 * cM * 4);
    float* bcat   = (float*)carve((size_t)2 * cM * 4);
    (void)in_sizes; (void)n_in; (void)out_size; (void)ws_size;

    auto nb = [](long total) { return (unsigned)((total + 255) / 256); };

    // ---- 1) weight conversions to padded bf16 ----
    cvt_pad<<<nb((long)1024 * 1024), 256, 0, stream>>>(W_hp,   Whp_bf,   1000, 1000, 1024, (long)1024 * 1024);
    cvt_pad<<<nb((long)1024 * 2016), 256, 0, stream>>>(W_op,   Wop_bf,   1000, 2000, 2016, (long)1024 * 2016);
    cvt_pad<<<nb((long)3008 * 2624), 256, 0, stream>>>(W_ih,   Wih_bf,   3000, 2620, 2624, (long)3008 * 2624);
    cvt_pad<<<nb((long)3008 * 1024), 256, 0, stream>>>(W_hh,   Whh_bf,   3000, 1000, 1024, (long)3008 * 1024);
    cvt_pad<<<nb((long)30016 * 512), 256, 0, stream>>>(W_pred, Wpred_bf, cV,   500,  512,  (long)30016 * 512);
    // concatenated maxout weight [W_lh | W_le | W_wv]
    zero_bf16<<<nb((long)1024 * 3648), 256, 0, stream>>>(Wcat_bf, (long)1024 * 3648);
    cvt_seg<<<nb((long)1000 * 1000), 256, 0, stream>>>(W_lh, Wcat_bf, 1000, 1000, 1000, 3648, 0);
    cvt_seg<<<nb((long)1000 * 620),  256, 0, stream>>>(W_le, Wcat_bf, 1000, 620,  620,  3648, 1000);
    cvt_seg<<<nb((long)1000 * 2000), 256, 0, stream>>>(W_wv, Wcat_bf, 1000, 2000, 2000, 3648, 1620);
    bias3_k<<<nb(2 * cM), 256, 0, stream>>>(b_lh, b_le, b_wv, bcat, 2 * cM);

    // ---- 2) activation conversions + embedding gather ----
    cvt_pad<<<nb((long)cB * 1024), 256, 0, stream>>>(hidden, hid_bf, cB, cH, 1024, (long)cB * 1024);
    cvt_pad<<<nb((long)cB * cS * 2016), 256, 0, stream>>>(outputs, out_bf, cB * cS, 2 * cH, 2016, (long)cB * cS * 2016);
    gather_emb<<<cB, 256, 0, stream>>>(x, emb, emb_x);

    dim3 wv(32);
    // ---- 3) attention GEMMs ----
    gemm_bf16_wmma<<<dim3(16, cB / 32),        wv, 0, stream>>>(hid_bf, Whp_bf, b_hp, h_proj, cB,      cA, 1024, cA);
    gemm_bf16_wmma<<<dim3(16, (cB * cS) / 32), wv, 0, stream>>>(out_bf, Wop_bf, b_op, o_proj, cB * cS, cA, 2016, cA);

    // ---- 4) energy / softmax / context ----
    energy_k<<<cB * cS, 256, 0, stream>>>(h_proj, o_proj, v_e, energy);
    attn_ctx_k<<<cB, 256, 0, stream>>>(energy, outputs, ctx);

    // ---- 5) build concatenated bf16 activations ----
    zero_bf16<<<nb((long)cB * 2624), 256, 0, stream>>>(gcat_bf, (long)cB * 2624);
    zero_bf16<<<nb((long)cB * 3648), 256, 0, stream>>>(acat_bf, (long)cB * 3648);
    cvt_seg<<<nb((long)cB * cE),     256, 0, stream>>>(emb_x,  gcat_bf, cB, cE,     cE,     2624, 0);
    cvt_seg<<<nb((long)cB * 2 * cH), 256, 0, stream>>>(ctx,    gcat_bf, cB, 2 * cH, 2 * cH, 2624, cE);
    cvt_seg<<<nb((long)cB * cH),     256, 0, stream>>>(hidden, acat_bf, cB, cH,     cH,     3648, 0);
    cvt_seg<<<nb((long)cB * cE),     256, 0, stream>>>(emb_x,  acat_bf, cB, cE,     cE,     3648, cH);
    cvt_seg<<<nb((long)cB * 2 * cH), 256, 0, stream>>>(ctx,    acat_bf, cB, 2 * cH, 2 * cH, 3648, cH + cE);

    // ---- 6) GRU GEMMs + gates ----
    gemm_bf16_wmma<<<dim3(47, cB / 32), wv, 0, stream>>>(gcat_bf, Wih_bf, b_ih, gi, cB, 3 * cH, 2624, 3 * cH);
    gemm_bf16_wmma<<<dim3(47, cB / 32), wv, 0, stream>>>(hid_bf,  Whh_bf, b_hh, gh, cB, 3 * cH, 1024, 3 * cH);
    gru_k<<<nb((long)cB * cH), 256, 0, stream>>>(gi, gh, hidden, h_new);

    // ---- 7) maxout head + vocabulary projection ----
    gemm_bf16_wmma<<<dim3(16, cB / 32), wv, 0, stream>>>(acat_bf, Wcat_bf, bcat, mo_in, cB, 2 * cM, 3648, 2 * cM);
    maxout_k<<<nb((long)cB * 512), 256, 0, stream>>>(mo_in, mo_bf);
    gemm_bf16_wmma<<<dim3(469, cB / 32), wv, 0, stream>>>(mo_bf, Wpred_bf, b_pred, pred, cB, cV, 512, cV);
}
